// TinyVanillaRNN_19344532701613
// MI455X (gfx1250) — matrix operand order repaired and checked by reading, not compiled, not run on hardware
//
#include <hip/hip_runtime.h>
#include <hip/hip_bf16.h>
#include <stdint.h>

// Problem sizes (fixed by the reference)
#define T_STEPS 16384
#define D_IN    512
#define H_DIM   1024
#define O_DIM   512

// Recurrence phase config
#define NWG         16                    // persistent workgroups (co-resident)
#define ROWS_PER_WG (H_DIM / NWG)         // 64 rows of W per workgroup
#define CHUNK       256                   // K elements per K-phase chunk
#define CHUNK_S     260                   // chunk stride (260 % 64 == 4 -> bank spread)
#define ROW_S       (4 * CHUNK_S)         // 1040 floats per padded W row
#define SMEM_FLOATS (ROWS_PER_WG * ROW_S + 4 * CHUNK_S)   // W slice + chunked h

typedef __attribute__((ext_vector_type(16))) __bf16 v16bf;
typedef __attribute__((ext_vector_type(8)))  float  v8f;

union FragBF { v16bf v; uint4 q[2]; };

// ---------------------------------------------------------------------------
// Phase A1: Ut[d][h] = U[h][d] + bh[h]   (transpose + fold bias; 2 MB, L2 resident)
// ---------------------------------------------------------------------------
__global__ void k_prep_ut(const float* __restrict__ U, const float* __restrict__ bh,
                          float* __restrict__ Ut) {
    int idx = blockIdx.x * blockDim.x + threadIdx.x;   // over D*H
    int d = idx >> 10;            // / H_DIM
    int h = idx & (H_DIM - 1);
    Ut[idx] = U[h * D_IN + d] + bh[h];
}

// ---------------------------------------------------------------------------
// Phase A2: Uxb[t][h] = Ut[x_t][h]   (coalesced row gather, 64 MB write)
// ---------------------------------------------------------------------------
__global__ void k_gather_ux(const float* __restrict__ Ut, const int* __restrict__ xs,
                            float* __restrict__ Uxb) {
    size_t idx = (size_t)blockIdx.x * blockDim.x + threadIdx.x;  // over T*H
    int t = (int)(idx >> 10);
    int h = (int)(idx & (H_DIM - 1));
    Uxb[idx] = Ut[(size_t)xs[t] * H_DIM + h];
}

// ---------------------------------------------------------------------------
// Phase A3: V -> bf16, and h <- h0
// ---------------------------------------------------------------------------
__global__ void k_prep_vb(const float* __restrict__ V, __hip_bfloat16* __restrict__ Vb,
                          const float* __restrict__ h0, float* __restrict__ hg) {
    int idx = blockIdx.x * blockDim.x + threadIdx.x;   // over O*H
    Vb[idx] = __float2bfloat16(V[idx]);
    if (idx < H_DIM) hg[idx] = h0[idx];
}

// ---------------------------------------------------------------------------
// Phase B: sequential recurrence. 16 persistent WGs, W slice LDS-resident
// (~266 KB of the 320 KB CDNA5 WGP LDS), per-step agent-scope barrier.
// Layout: W row r stored as 4 chunks of 256 floats, chunk stride 260.
//   b128 word index mod 64 = 16*row + 4*c + 4*jb + e -> 2 words/bank (optimal)
//   h chunk reads: banks (4c + 4jb + e) mod 64 -> broadcast, conflict-free
// h is staged into LDS each step via GLOBAL_LOAD_ASYNC_TO_LDS_B128 (ASYNCcnt).
// ---------------------------------------------------------------------------
__global__ void __launch_bounds__(256)
k_rnn(const float* __restrict__ W, const float* __restrict__ Uxb,
      float* __restrict__ hg, __hip_bfloat16* __restrict__ hsb,
      unsigned* __restrict__ bar) {
    extern __shared__ float smem[];
    float* Wl = smem;                               // ROWS_PER_WG * ROW_S
    float* hl = smem + ROWS_PER_WG * ROW_S;         // 4 * CHUNK_S chunked h staging

    const int tid = threadIdx.x;
    const int g   = blockIdx.x;

    // Load this WG's 64x1024 fp32 slice of W into chunk-padded LDS (once).
    for (int i = tid; i < ROWS_PER_WG * H_DIM; i += 256) {
        int r = i >> 10;
        int k = i & (H_DIM - 1);
        Wl[r * ROW_S + (k >> 8) * CHUNK_S + (k & (CHUNK - 1))] =
            W[(size_t)(g * ROWS_PER_WG + r) * H_DIM + k];
    }
    const int row  = tid >> 2;          // 0..63
    const int c    = tid & 3;           // K phase 0..3
    const int grow = g * ROWS_PER_WG + row;
    const float* wbase = &Wl[row * ROW_S + c * CHUNK_S];
    const float* hbase = &hl[c * CHUNK_S];

    // Per-lane async-copy operands: lane copies h[tid*4 .. tid*4+3] into the
    // chunked staging buffer. Low 32 bits of a generic LDS pointer == LDS byte
    // offset (flat-aperture mapping), which is what the async VDST field wants.
    const unsigned lds_dst =
        (unsigned)(uintptr_t)(hl + ((tid >> 6) * CHUNK_S + ((tid * 4) & (CHUNK - 1))));
    const unsigned long long gsrc = (unsigned long long)(uintptr_t)(hg + tid * 4);
    __syncthreads();

    for (int t = 0; t < T_STEPS; ++t) {
        // Stage current h (4 KB) into LDS via the async copy engine, DEV scope
        // so it reads through to L2 and sees peer workgroups' stores.
        asm volatile("global_load_async_to_lds_b128 %0, %1, off scope:SCOPE_DEV"
                     :: "v"(lds_dst), "v"(gsrc) : "memory");
        asm volatile("s_wait_asynccnt 0x0" ::: "memory");
        __syncthreads();

        float acc = 0.f;
#pragma unroll 4
        for (int jb = 0; jb < CHUNK; jb += 4) {
            float4 w  = *reinterpret_cast<const float4*>(wbase + jb);  // ds_load_b128
            float4 hv = *reinterpret_cast<const float4*>(hbase + jb);  // ds_load_b128
            acc = __builtin_fmaf(w.x, hv.x, acc);
            acc = __builtin_fmaf(w.y, hv.y, acc);
            acc = __builtin_fmaf(w.z, hv.z, acc);
            acc = __builtin_fmaf(w.w, hv.w, acc);
        }

        // Reduce the 4 K-phases of each row (lanes differ in low 2 bits).
        acc += __shfl_xor(acc, 1, 32);
        acc += __shfl_xor(acc, 2, 32);

        if (c == 0) {
            float hn = tanhf(acc + Uxb[(size_t)t * H_DIM + grow]);  // bh folded into Uxb
            hg[grow] = hn;                                          // fp32 for recurrence
            hsb[(size_t)t * H_DIM + grow] = __float2bfloat16(hn);   // bf16 for WMMA GEMM
        }
        __syncthreads();   // all waves' stores issued before arrive

        if (tid == 0) {
            __hip_atomic_fetch_add(&bar[t], 1u, __ATOMIC_RELEASE, __HIP_MEMORY_SCOPE_AGENT);
            while (__hip_atomic_load(&bar[t], __ATOMIC_ACQUIRE, __HIP_MEMORY_SCOPE_AGENT) < NWG)
                __builtin_amdgcn_s_sleep(2);
        }
        __syncthreads();
        __builtin_amdgcn_fence(__ATOMIC_ACQUIRE, "agent");  // invalidate WGP$ before next step
    }
}

// ---------------------------------------------------------------------------
// Phase C: logits = hs(bf16) @ V(bf16)^T + by, fp32 accumulate via
// v_wmma_f32_16x16x32_bf16. Each wave: one 16x64 output strip (4 acc tiles).
// ---------------------------------------------------------------------------
__global__ void __launch_bounds__(256)
k_gemm(const __hip_bfloat16* __restrict__ A,   // hs  : T x H row-major
       const __hip_bfloat16* __restrict__ B,   // Vb  : O x H row-major (= B^T column store)
       const float* __restrict__ by,
       float* __restrict__ out) {              // T x O
    const int lane = threadIdx.x & 31;
    const int wave = threadIdx.x >> 5;
    const int id   = blockIdx.x * 8 + wave;    // 8192 wave-tasks
    const int mt   = id >> 3;                  // 0..1023  (16 rows each; shared per block)
    const int nt   = id & 7;                   // 0..7     (64 cols each)
    const int m    = lane & 15;
    const int half = lane >> 4;

    const size_t arow = (size_t)(mt * 16 + m) * H_DIM;

    v8f acc[4] = {v8f{}, v8f{}, v8f{}, v8f{}};
    float bbias[4];
#pragma unroll
    for (int j = 0; j < 4; ++j) bbias[j] = by[nt * 64 + j * 16 + m];

    for (int kk = 0; kk < H_DIM; kk += 32) {
        // A fragment (16x32 bf16): lane<16 holds K[0..8)+K[16..24), lane>=16 K[8..16)+K[24..32)
        FragBF a;
        a.q[0] = *(const uint4*)(A + arow + kk + half * 8);
        a.q[1] = *(const uint4*)(A + arow + kk + 16 + half * 8);
        __builtin_prefetch(A + arow + kk + 64, 0, 3);   // global_prefetch_b8, near-resident

        FragBF b[4];
#pragma unroll
        for (int j = 0; j < 4; ++j) {
            // B fragment (32x16): lane = column n, lane<16 holds K[0..8)+K[16..24), lane>=16 K[8..16)+K[24..32)
            const __hip_bfloat16* bp = B + (size_t)(nt * 64 + j * 16 + m) * H_DIM + kk + half * 8;
            b[j].q[0] = *(const uint4*)(bp);
            b[j].q[1] = *(const uint4*)(bp + 16);
        }
#pragma unroll
        for (int j = 0; j < 4; ++j)
            acc[j] = __builtin_amdgcn_wmma_f32_16x16x32_bf16(
                false, a.v, false, b[j].v, (short)0, acc[j], false, false);
    }

    // D layout: VGPR v -> M = v + 8*half, N = lane&15
#pragma unroll
    for (int v = 0; v < 8; ++v) {
        float* orow = out + (size_t)(mt * 16 + v + 8 * half) * O_DIM + nt * 64 + m;
#pragma unroll
        for (int j = 0; j < 4; ++j)
            orow[j * 16] = acc[j][v] + bbias[j];
    }
}

// ---------------------------------------------------------------------------
// Final: copy h_T into the output tail
// ---------------------------------------------------------------------------
__global__ void k_copy_ht(const float* __restrict__ hg, float* __restrict__ outh) {
    int i = blockIdx.x * blockDim.x + threadIdx.x;
    if (i < H_DIM) outh[i] = hg[i];
}

// ---------------------------------------------------------------------------
extern "C" void kernel_launch(void* const* d_in, const int* in_sizes, int n_in,
                              void* d_out, int out_size, void* d_ws, size_t ws_size,
                              hipStream_t stream) {
    const int*   xs = (const int*)d_in[0];
    const float* h0 = (const float*)d_in[1];
    const float* U  = (const float*)d_in[2];
    const float* W  = (const float*)d_in[3];
    const float* V  = (const float*)d_in[4];
    const float* bh = (const float*)d_in[5];
    const float* by = (const float*)d_in[6];

    char* ws = (char*)d_ws;
    float*          Uxb = (float*)(ws);                                   // 64 MB
    __hip_bfloat16* hsb = (__hip_bfloat16*)(ws + ((size_t)64 << 20));     // 32 MB
    float*          Ut  = (float*)(ws + ((size_t)96 << 20));              //  2 MB
    __hip_bfloat16* Vb  = (__hip_bfloat16*)(ws + ((size_t)98 << 20));     //  1 MB
    float*          hg  = (float*)(ws + ((size_t)99 << 20));              //  4 KB
    unsigned*       bar = (unsigned*)(ws + ((size_t)99 << 20) + (1 << 16)); // 64 KB

    float* logits = (float*)d_out;
    float* hT     = logits + (size_t)T_STEPS * O_DIM;

    // Re-zero per-step barrier counters every call (capture-legal memset node).
    hipMemsetAsync(bar, 0, T_STEPS * sizeof(unsigned), stream);

    k_prep_ut<<<(D_IN * H_DIM) / 256, 256, 0, stream>>>(U, bh, Ut);
    k_gather_ux<<<(int)(((size_t)T_STEPS * H_DIM) / 256), 256, 0, stream>>>(Ut, xs, Uxb);
    k_prep_vb<<<(O_DIM * H_DIM) / 256, 256, 0, stream>>>(V, Vb, h0, hg);

    size_t smem = (size_t)SMEM_FLOATS * sizeof(float);   // ~270 KB < 320 KB
    (void)hipFuncSetAttribute((const void*)k_rnn,
                              hipFuncAttributeMaxDynamicSharedMemorySize, (int)smem);
    k_rnn<<<NWG, 256, smem, stream>>>(W, Uxb, hg, hsb, bar);

    k_gemm<<<(T_STEPS / 16) * (O_DIM / 64) / 8, 256, 0, stream>>>(hsb, Vb, by, logits);
    k_copy_ht<<<(H_DIM + 255) / 256, 256, 0, stream>>>(hg, hT);
}
